// PWCNet_76132590289308
// MI455X (gfx1250) — matrix operand order repaired and checked by reading, not compiled, not run on hardware
//
#include <hip/hip_runtime.h>

typedef __attribute__((ext_vector_type(16))) _Float16 v16h;
typedef __attribute__((ext_vector_type(8)))  _Float16 v8h;
typedef __attribute__((ext_vector_type(8)))  float    v8f;

__device__ __forceinline__ int imin(int a, int b) { return a < b ? a : b; }
__device__ __forceinline__ int imax(int a, int b) { return a > b ? a : b; }

#define NEG_SLOPE 0.1f

// All f16 activations live in NHWC with a 1-pixel zero halo:
//   layout [B, H+2, W+2, CP], interior pixel (y,x) at ((b*(H+2)+y+1)*(W+2)+x+1)*CP.

// ---------------------------------------------------------------------------
// Weight pack: f32 OIHW -> f16 [CoutP][9*CinP], K order k = r*CinP + c,
// zero-padded in both Cout and Cin so the conv kernel loads unconditionally.
// ---------------------------------------------------------------------------
__global__ void k_pack_w(const float* __restrict__ w, _Float16* __restrict__ wp,
                         int Cout, int Cin, int CoutP, int CinP) {
  const int Krow = 9 * CinP;
  int i = blockIdx.x * blockDim.x + threadIdx.x;
  if (i >= CoutP * Krow) return;
  int m = i / Krow, k = i - m * Krow;
  int r = k / CinP, c = k - r * CinP;
  _Float16 v = (_Float16)0.0f;
  if (m < Cout && c < Cin) v = (_Float16)w[((size_t)m * Cin + c) * 9 + r];
  wp[i] = v;
}

// Zero the 1-pixel halo ring of an NHWC-halo tensor.
__global__ void k_border_zero(_Float16* __restrict__ d, int Bn, int H, int W, int CP) {
  const int Hp = H + 2, Wp = W + 2;
  const int ring = 2 * Wp + 2 * H;             // halo pixels per image
  int i = blockIdx.x * blockDim.x + threadIdx.x;
  int total = Bn * ring * CP;
  if (i >= total) return;
  int c = i % CP; int t = i / CP;
  int rp = t % ring; int b = t / ring;
  int y, x;
  if (rp < Wp)            { y = 0;       x = rp; }
  else if (rp < 2 * Wp)   { y = Hp - 1;  x = rp - Wp; }
  else { int q = rp - 2 * Wp; y = 1 + (q >> 1); x = (q & 1) ? (Wp - 1) : 0; }
  d[((size_t)(b * Hp + y) * Wp + x) * CP + c] = (_Float16)0.0f;
}

// f32 NCHW -> f16 NHWC-halo interior, channel padding (zeros beyond Cs).
__global__ void k_nchw_to_nhwc16(const float* __restrict__ s, _Float16* __restrict__ d,
                                 int Bn, int Cs, int CP, int H, int W) {
  int i = blockIdx.x * blockDim.x + threadIdx.x;
  int total = Bn * H * W * CP;
  if (i >= total) return;
  int c = i % CP; int pix = i / CP;
  int x = pix % W; int t = pix / W; int y = t % H; int b = t / H;
  const size_t o = ((size_t)(b * (H + 2) + y + 1) * (W + 2) + x + 1) * CP + c;
  d[o] = (c < Cs) ? (_Float16)s[(((size_t)b * Cs + c) * H + y) * W + x] : (_Float16)0.0f;
}

__global__ void k_fill_f32(float* __restrict__ p, float v, int n) {
  int i = blockIdx.x * blockDim.x + threadIdx.x;
  if (i < n) p[i] = v;
}

// Copy Cs channels between NHWC-halo tensors of identical geometry (whole
// halo grid; source halo is zero so destination halo channels become zero).
__global__ void k_chancopy(const _Float16* __restrict__ s, _Float16* __restrict__ d,
                           int total /*Bn*Hp*Wp*Cs*/, int Cs, int CsP, int CdP, int co) {
  int i = blockIdx.x * blockDim.x + threadIdx.x;
  if (i >= total) return;
  int c = i % Cs; int pix = i / Cs;
  d[(size_t)pix * CdP + co + c] = s[(size_t)pix * CsP + c];
}

// Cast f32 NCHW flow (2ch) into NHWC-halo concat tensor interior at offset co.
__global__ void k_flow_to_cat(const float* __restrict__ flow, _Float16* __restrict__ d,
                              int Bn, int Hh, int Ww, int CdP, int co) {
  int i = blockIdx.x * blockDim.x + threadIdx.x;
  int total = Bn * Hh * Ww * 2;
  if (i >= total) return;
  int c = i & 1; int pix = i >> 1;
  int x = pix % Ww; int t = pix / Ww; int y = t % Hh; int b = t / Hh;
  const size_t o = ((size_t)(b * (Hh + 2) + y + 1) * (Ww + 2) + x + 1) * CdP + co + c;
  d[o] = (_Float16)flow[(((size_t)b * 2 + c) * Hh + y) * Ww + x];
}

// Zero channels [c0, c0+nc) over the whole halo grid of an NHWC-halo tensor.
__global__ void k_zero_chans(_Float16* __restrict__ d, int total /*Bn*Hp*Wp*nc*/,
                             int nc, int CdP, int c0) {
  int i = blockIdx.x * blockDim.x + threadIdx.x;
  if (i >= total) return;
  int c = i % nc; int pix = i / nc;
  d[(size_t)pix * CdP + c0 + c] = (_Float16)0.0f;
}

// ---------------------------------------------------------------------------
// 3x3 conv (pad 1, stride 1|2) as implicit GEMM on v_wmma_f32_16x16x32_f16.
// Halo layout => every tap address in-bounds and halo data is zero, so the
// B fragment is two UNCONDITIONAL b128 loads (no masking at all).
// Each wave: 16 Cout x 4x16 pixels (4 accumulators). 8 waves per block.
// ---------------------------------------------------------------------------
#define NT 4
__global__ __launch_bounds__(256)
void k_conv3x3_wmma(const _Float16* __restrict__ src,   // halo NHWC [B,H+2,W+2,CinP]
                    const _Float16* __restrict__ wp,    // [CoutP][9*CinP]
                    const float* __restrict__ bias,
                    _Float16* __restrict__ dst16,       // halo NHWC [B,Ho+2,Wo+2,CoutP] or null
                    const float* __restrict__ addsrc,   // NCHW flow or null
                    float* __restrict__ dst32,          // NCHW flow-out or null
                    int Bn, int CinP, int H, int W,
                    int Cout, int CoutP, int Ho, int Wo,
                    int stride, int doAct)
{
  const int lane = threadIdx.x & 31;
  const int wv   = threadIdx.x >> 5;
  const int totalPix = Bn * Ho * Wo;
  const int pixBase  = (blockIdx.x * 8 + wv) * (16 * NT);
  if (pixBase >= totalPix) return;          // wave-uniform: EXEC all-1s for WMMA

  const int grp = lane >> 4;                // lane group 0 / 1
  const int nn  = lane & 15;
  const int coutBase = blockIdx.y * 16;
  const int Hp = H + 2, Wp = W + 2;

  int  pp[NT], oxv[NT], oyv[NT], bbv[NT];
  bool pv[NT];
#pragma unroll
  for (int t = 0; t < NT; ++t) {
    const int p = pixBase + 16 * t + nn;
    pv[t] = p < totalPix;
    const int pc = pv[t] ? p : 0;           // clamp: reads valid memory, result unused
    pp[t]  = pc;
    oxv[t] = pc % Wo;
    const int ty = pc / Wo;
    oyv[t] = ty % Ho;
    bbv[t] = ty / Ho;
  }

  // Input base pointer of each tile at tap (ky=0,kx=0) (halo removes the -1).
  const _Float16* sb[NT];
#pragma unroll
  for (int t = 0; t < NT; ++t)
    sb[t] = src + (((size_t)bbv[t] * Hp + oyv[t] * stride) * Wp + oxv[t] * stride) * CinP + grp * 8;

  const size_t Krow = (size_t)9 * CinP;
  // A row: M = lane&15 both lane groups; halves at K offsets grp*8 + {0..7}
  // and 16 + grp*8 + {0..7} within each 32-chunk.
  const _Float16* arow = wp + (size_t)(coutBase + nn) * Krow + grp * 8;

  v8f acc[NT];
#pragma unroll
  for (int t = 0; t < NT; ++t) acc[t] = (v8f){};

  for (int r = 0; r < 9; ++r) {
    const int ky = r / 3;
    const int kx = r - ky * 3;
    const size_t tapOff = ((size_t)ky * Wp + kx) * CinP;
    const _Float16* pa = arow + (size_t)r * CinP;

    for (int c0 = 0; c0 < CinP; c0 += 32) {
      const v8h a_lo = *(const v8h*)(pa + c0);
      const v8h a_hi = *(const v8h*)(pa + c0 + 16);
      v16h a;
#pragma unroll
      for (int j = 0; j < 8; ++j) { a[j] = a_lo[j]; a[8 + j] = a_hi[j]; }

#pragma unroll
      for (int t = 0; t < NT; ++t) {
        const _Float16* sp = sb[t] + tapOff + c0;
        const v8h ql = *(const v8h*)(sp);
        const v8h qh = *(const v8h*)(sp + 16);
        v16h b;
#pragma unroll
        for (int j = 0; j < 8; ++j) { b[j] = ql[j]; b[8 + j] = qh[j]; }
        acc[t] = __builtin_amdgcn_wmma_f32_16x16x32_f16(false, a, false, b,
                                                        (short)0, acc[t], false, false);
      }
    }
  }

  // Epilogue. D layout: VGPR r -> M = r + 8*grp, N = lane&15: this lane owns
  // 8 consecutive output channels at its pixel -> one 16-B NHWC store.
  const int chBase = coutBase + 8 * grp;
  const int Hop = Ho + 2, Wop = Wo + 2;
  float bv[8];
#pragma unroll
  for (int r = 0; r < 8; ++r) {
    const int ch = chBase + r;
    bv[r] = (ch < Cout) ? bias[ch] : 0.0f;
  }

#pragma unroll
  for (int t = 0; t < NT; ++t) {
    if (!pv[t]) continue;
    if (dst16) {
      v8h o;
#pragma unroll
      for (int r = 0; r < 8; ++r) {
        const int ch = chBase + r;
        float v = (ch < Cout) ? acc[t][r] + bv[r] : 0.0f;
        if (doAct) v = v > 0.0f ? v : NEG_SLOPE * v;
        o[r] = (_Float16)v;
      }
      const size_t oidx = ((size_t)(bbv[t] * Hop + oyv[t] + 1) * Wop + oxv[t] + 1) * CoutP + chBase;
      *(v8h*)(dst16 + oidx) = o;
    }
    if (dst32) {
#pragma unroll
      for (int r = 0; r < 8; ++r) {
        const int ch = chBase + r;
        if (ch < Cout) {
          float v = acc[t][r] + bv[r];
          if (doAct) v = v > 0.0f ? v : NEG_SLOPE * v;
          const size_t oidx = (((size_t)bbv[t] * Cout + ch) * Ho + oyv[t]) * Wo + oxv[t];
          dst32[oidx] = (addsrc ? addsrc[oidx] : 0.0f) + v;
        }
      }
    }
  }
}

// ---------------------------------------------------------------------------
// Cost volume (NHWC-halo): cv[d] = mean_c f1[y,x,c]*f2w[y-dy,x-dx,c],
// d=(dx+4)*9+(dy+4). Vectorized over padded channels CP (pad is zero).
// ---------------------------------------------------------------------------
__global__ void k_costvol(const _Float16* __restrict__ f1, const _Float16* __restrict__ f2,
                          _Float16* __restrict__ xcat, int Bn, int C, int CP,
                          int Hh, int Ww, int CxP)
{
  int i = blockIdx.x * blockDim.x + threadIdx.x;
  int total = Bn * 81 * Hh * Ww;
  if (i >= total) return;
  int x = i % Ww; int t = i / Ww; int y = t % Hh; t /= Hh; int d = t % 81; int b = t / 81;
  int dx = d / 9 - 4;
  int dy = d - (d / 9) * 9 - 4;
  int sy = y - dy, sx = x - dx;
  const int Hp = Hh + 2, Wpp = Ww + 2;
  float s = 0.0f;
  if (sy >= 0 && sy < Hh && sx >= 0 && sx < Ww) {
    const _Float16* p1 = f1 + ((size_t)(b * Hp + y + 1) * Wpp + x + 1) * CP;
    const _Float16* p2 = f2 + ((size_t)(b * Hp + sy + 1) * Wpp + sx + 1) * CP;
    for (int c = 0; c < CP; c += 8) {
      const v8h a  = *(const v8h*)(p1 + c);
      const v8h bb = *(const v8h*)(p2 + c);
#pragma unroll
      for (int j = 0; j < 8; ++j) s += (float)a[j] * (float)bb[j];
    }
  }
  xcat[((size_t)(b * Hp + y + 1) * Wpp + x + 1) * CxP + d] = (_Float16)(s / (float)C);
}

// ---------------------------------------------------------------------------
// Bilinear warp (NHWC-halo features, NCHW f32 flow), zero outside.
// ---------------------------------------------------------------------------
__global__ void k_warp(const _Float16* __restrict__ feat, const float* __restrict__ flow,
                       _Float16* __restrict__ out, int Bn, int CP, int Hh, int Ww)
{
  int i = blockIdx.x * blockDim.x + threadIdx.x;
  int total = Bn * Hh * Ww * CP;
  if (i >= total) return;
  int c = i % CP; int pix = i / CP;
  int x = pix % Ww; int t = pix / Ww; int y = t % Hh; int b = t / Hh;
  const int HW = Hh * Ww, Hp = Hh + 2, Wpp = Ww + 2;
  const float fx = flow[((size_t)b * 2 + 0) * HW + (size_t)y * Ww + x];
  const float fy = flow[((size_t)b * 2 + 1) * HW + (size_t)y * Ww + x];
  const float xs = (float)x + fx, ys = (float)y + fy;
  const float x0f = floorf(xs), y0f = floorf(ys);
  const float wx = xs - x0f, wy = ys - y0f;
  const int x0 = (int)x0f, y0 = (int)y0f;
  float acc = 0.0f;
#pragma unroll
  for (int dyy = 0; dyy < 2; ++dyy)
#pragma unroll
    for (int dxx = 0; dxx < 2; ++dxx) {
      const int xi = x0 + dxx, yi = y0 + dyy;
      const float wt = (dxx ? wx : 1.0f - wx) * (dyy ? wy : 1.0f - wy);
      const bool valid = (xi >= 0) && (xi < Ww) && (yi >= 0) && (yi < Hh);
      const int xc = imin(imax(xi, 0), Ww - 1);
      const int yc = imin(imax(yi, 0), Hh - 1);
      acc += (float)feat[((size_t)(b * Hp + yc + 1) * Wpp + xc + 1) * CP + c] * wt * (valid ? 1.0f : 0.0f);
    }
  out[((size_t)(b * Hp + y + 1) * Wpp + x + 1) * CP + c] = (_Float16)acc;
}

// ---------------------------------------------------------------------------
// Half-pixel bilinear resize (f32 NCHW) with scalar multiplier.
// ---------------------------------------------------------------------------
__global__ void k_upsample(const float* __restrict__ s, float* __restrict__ d,
                           int Bn, int C, int Hs, int Ws, int Hd, int Wd, float mul)
{
  int i = blockIdx.x * blockDim.x + threadIdx.x;
  int total = Bn * C * Hd * Wd;
  if (i >= total) return;
  int X = i % Wd; int t = i / Wd; int Y = t % Hd; t /= Hd; int c = t % C; int b = t / C;
  float fy = ((float)Y + 0.5f) * (float)Hs / (float)Hd - 0.5f;
  float fx = ((float)X + 0.5f) * (float)Ws / (float)Wd - 0.5f;
  fy = fminf(fmaxf(fy, 0.0f), (float)(Hs - 1));
  fx = fminf(fmaxf(fx, 0.0f), (float)(Ws - 1));
  const int y0 = (int)fy, x0 = (int)fx;
  const int y1 = imin(y0 + 1, Hs - 1), x1 = imin(x0 + 1, Ws - 1);
  const float wy = fy - (float)y0, wx = fx - (float)x0;
  const float* base = s + ((size_t)b * C + c) * Hs * Ws;
  const float v = (1.0f - wy) * ((1.0f - wx) * base[(size_t)y0 * Ws + x0] + wx * base[(size_t)y0 * Ws + x1])
                +          wy * ((1.0f - wx) * base[(size_t)y1 * Ws + x0] + wx * base[(size_t)y1 * Ws + x1]);
  d[i] = v * mul;
}

// ---------------------------------------------------------------------------
// Host orchestration
// ---------------------------------------------------------------------------
extern "C" void kernel_launch(void* const* d_in, const int* in_sizes, int n_in,
                              void* d_out, int out_size, void* d_ws, size_t ws_size,
                              hipStream_t stream)
{
  (void)in_sizes; (void)n_in; (void)out_size; (void)ws_size;
  const int Bn = 2, Himg = 384, Wimg = 512;

  const float* img1 = (const float*)d_in[0];
  const float* img2 = (const float*)d_in[1];

  auto W_bb = [&](int i)        { return (const float*)d_in[2 + 2 * i]; };
  auto B_bb = [&](int i)        { return (const float*)d_in[3 + 2 * i]; };
  auto W_fe = [&](int l, int j) { return (const float*)d_in[26 + l * 12 + 2 * j]; };
  auto B_fe = [&](int l, int j) { return (const float*)d_in[27 + l * 12 + 2 * j]; };
  auto W_cx = [&](int l, int j) { return (const float*)d_in[98 + l * 14 + 2 * j]; };
  auto B_cx = [&](int l, int j) { return (const float*)d_in[99 + l * 14 + 2 * j]; };

  const int bb_cin[12]  = {3, 16, 16, 32, 32, 64, 64, 96, 96, 128, 128, 196};
  const int bb_cout[12] = {16, 16, 32, 32, 64, 64, 96, 96, 128, 128, 196, 196};
  const int feat_ch[6]  = {196, 128, 96, 64, 32, 16};
  const int lvl_h[6]    = {6, 12, 24, 48, 96, 192};
  const int lvl_w[6]    = {8, 16, 32, 64, 128, 256};

  auto padc = [](int c) { return (c + 31) & ~31; };
  auto haloN = [&](int H, int W, int CP) { return (size_t)Bn * (H + 2) * (W + 2) * CP; };

  char* wsp = (char*)d_ws;
  size_t off = 0;
  auto alloc = [&](size_t bytes) -> void* {
    void* p = wsp + off;
    off += (bytes + 255) & ~(size_t)255;
    return p;
  };

  auto borderZero = [&](_Float16* d, int H, int W, int CP) {
    const int ring = 2 * (W + 2) + 2 * H;
    const int n = Bn * ring * CP;
    k_border_zero<<<(n + 255) / 256, 256, 0, stream>>>(d, Bn, H, W, CP);
  };

  // ---- pack all weights (every call: deterministic) ----
  auto pack = [&](const float* w, int Cout, int Cin) -> _Float16* {
    const int CoutP = padc(Cout), CinP = padc(Cin);
    const int total = CoutP * 9 * CinP;
    _Float16* p = (_Float16*)alloc((size_t)total * sizeof(_Float16));
    k_pack_w<<<(total + 255) / 256, 256, 0, stream>>>(w, p, Cout, Cin, CoutP, CinP);
    return p;
  };

  _Float16* bbW[12];
  _Float16* feW[6][6];
  _Float16* cxW[6][7];
  for (int i = 0; i < 12; ++i) bbW[i] = pack(W_bb(i), bb_cout[i], bb_cin[i]);
  for (int l = 0; l < 6; ++l) {
    const int fc = feat_ch[l];
    const int fe_cin[6]  = {81 + fc + 2, 128, 128, 96, 64, 32};
    const int fe_cout[6] = {128, 128, 96, 64, 32, 2};
    for (int j = 0; j < 6; ++j) feW[l][j] = pack(W_fe(l, j), fe_cout[j], fe_cin[j]);
    const int cx_cin[7]  = {2, 128, 128, 128, 96, 64, 32};
    const int cx_cout[7] = {128, 128, 128, 96, 64, 32, 2};
    for (int j = 0; j < 7; ++j) cxW[l][j] = pack(W_cx(l, j), cx_cout[j], cx_cin[j]);
  }

  // ---- activation buffers (NHWC-halo f16) ----
  _Float16* img1h = (_Float16*)alloc(haloN(Himg, Wimg, 32) * sizeof(_Float16));
  _Float16* img2h = (_Float16*)alloc(haloN(Himg, Wimg, 32) * sizeof(_Float16));
  _Float16 *f1s[6], *f2s[6];
  for (int l = 0; l < 6; ++l) {
    const size_t n = haloN(lvl_h[l], lvl_w[l], padc(feat_ch[l]));
    f1s[l] = (_Float16*)alloc(n * sizeof(_Float16));
    f2s[l] = (_Float16*)alloc(n * sizeof(_Float16));
  }
  _Float16* tmp    = (_Float16*)alloc(haloN(192, 256, 32)  * sizeof(_Float16));
  _Float16* f2wbuf = (_Float16*)alloc(haloN(192, 256, 32)  * sizeof(_Float16));
  _Float16* ping   = (_Float16*)alloc(haloN(192, 256, 128) * sizeof(_Float16));
  _Float16* pong   = (_Float16*)alloc(haloN(192, 256, 128) * sizeof(_Float16));
  _Float16* xcat   = (_Float16*)alloc(haloN(192, 256, 128) * sizeof(_Float16));
  _Float16* flow16 = (_Float16*)alloc(haloN(192, 256, 32)  * sizeof(_Float16));
  float* flowA = (float*)alloc((size_t)Bn * 2 * 192 * 256 * sizeof(float));
  float* flowB = (float*)alloc((size_t)Bn * 2 * 192 * 256 * sizeof(float));

  auto conv = [&](const _Float16* src, _Float16* wpk, const float* bias,
                  _Float16* dst16, const float* addsrc, float* dst32,
                  int CinP, int H, int W, int Cout, int CoutP, int stride, int doAct) {
    const int Ho = (H - 1) / stride + 1, Wo = (W - 1) / stride + 1;
    if (dst16) borderZero(dst16, Ho, Wo, CoutP);   // consumer reads a zero halo
    const int totalPix = Bn * Ho * Wo;
    const int pixPerBlk = 8 * 16 * NT;
    dim3 g((totalPix + pixPerBlk - 1) / pixPerBlk, CoutP / 16, 1);
    k_conv3x3_wmma<<<g, 256, 0, stream>>>(src, wpk, bias, dst16, addsrc, dst32,
                                          Bn, CinP, H, W, Cout, CoutP, Ho, Wo,
                                          stride, doAct);
  };

  // ---- images -> NHWC-halo f16 (3 -> 32 padded channels) ----
  {
    borderZero(img1h, Himg, Wimg, 32);
    borderZero(img2h, Himg, Wimg, 32);
    const int n = Bn * Himg * Wimg * 32;
    k_nchw_to_nhwc16<<<(n + 255) / 256, 256, 0, stream>>>(img1, img1h, Bn, 3, 32, Himg, Wimg);
    k_nchw_to_nhwc16<<<(n + 255) / 256, 256, 0, stream>>>(img2, img2h, Bn, 3, 32, Himg, Wimg);
  }

  // ---- feature pyramids (feats stored coarse-first) ----
  auto pyramid = [&](const _Float16* imgh, _Float16** feats) {
    const _Float16* x = imgh;
    int H = Himg, Wd = Wimg, CinP = 32;
    for (int i = 0; i < 6; ++i) {
      const int Ca = bb_cout[2 * i], Cb = bb_cout[2 * i + 1];
      const int CaP = padc(Ca), CbP = padc(Cb);
      const int Ho = (H - 1) / 2 + 1, Wo = (Wd - 1) / 2 + 1;
      conv(x,   bbW[2 * i],     B_bb(2 * i),     tmp,          nullptr, nullptr,
           CinP, H, Wd, Ca, CaP, 2, 1);
      conv(tmp, bbW[2 * i + 1], B_bb(2 * i + 1), feats[5 - i], nullptr, nullptr,
           CaP, Ho, Wo, Cb, CbP, 1, 1);
      x = feats[5 - i]; H = Ho; Wd = Wo; CinP = CbP;
    }
  };
  pyramid(img1h, f1s);
  pyramid(img2h, f2s);

  // ---- coarse-to-fine flow loop ----
  float* flowCur = flowA;
  float* flowNxt = flowB;
  for (int lvl = 0; lvl < 6; ++lvl) {
    const int h = lvl_h[lvl], w = lvl_w[lvl], fc = feat_ch[lvl];
    const int HW = h * w, fcP = padc(fc);
    const int Cx = 81 + fc + 2, CxP = padc(Cx);
    const int haloPix = Bn * (h + 2) * (w + 2);
    const _Float16* warped;

    if (lvl == 0) {
      const int n = Bn * 2 * HW;
      k_fill_f32<<<(n + 255) / 256, 256, 0, stream>>>(flowCur, 0.0f, n);
      warped = f2s[0];
    } else {
      const int n = Bn * 2 * HW;
      k_upsample<<<(n + 255) / 256, 256, 0, stream>>>(flowCur, flowNxt,
                                                      Bn, 2, h / 2, w / 2, h, w, 2.0f);
      { float* t = flowCur; flowCur = flowNxt; flowNxt = t; }
      borderZero(f2wbuf, h, w, fcP);
      const int nw = Bn * HW * fcP;
      k_warp<<<(nw + 255) / 256, 256, 0, stream>>>(f2s[lvl], flowCur, f2wbuf, Bn, fcP, h, w);
      warped = f2wbuf;
    }

    // concat x = [cv(81) | f1(fc) | flow(2) | zero-pad] (NHWC-halo, CxP ch.)
    {
      borderZero(xcat, h, w, CxP);
      const int n1 = Bn * 81 * HW;
      k_costvol<<<(n1 + 255) / 256, 256, 0, stream>>>(f1s[lvl], warped, xcat,
                                                      Bn, fc, fcP, h, w, CxP);
      const int n2 = haloPix * fc;
      k_chancopy<<<(n2 + 255) / 256, 256, 0, stream>>>(f1s[lvl], xcat, n2, fc, fcP, CxP, 81);
      const int n3 = Bn * HW * 2;
      k_flow_to_cat<<<(n3 + 255) / 256, 256, 0, stream>>>(flowCur, xcat, Bn, h, w, CxP, 81 + fc);
      const int nc = CxP - Cx;
      if (nc > 0) {
        const int n4 = haloPix * nc;
        k_zero_chans<<<(n4 + 255) / 256, 256, 0, stream>>>(xcat, n4, nc, CxP, Cx);
      }
    }

    // flow estimation head
    {
      const int fe_cinP[6]  = {CxP, 128, 128, 96, 64, 32};
      const int fe_cout[6]  = {128, 128, 96, 64, 32, 2};
      const _Float16* cur = xcat;
      _Float16* bufs[2] = {ping, pong};
      for (int j = 0; j < 5; ++j) {
        conv(cur, feW[lvl][j], B_fe(lvl, j), bufs[j & 1], nullptr, nullptr,
             fe_cinP[j], h, w, fe_cout[j], padc(fe_cout[j]), 1, 1);
        cur = bufs[j & 1];
      }
      conv(cur, feW[lvl][5], B_fe(lvl, 5), nullptr, flowCur, flowNxt,
           32, h, w, 2, 32, 1, 0);
      { float* t = flowCur; flowCur = flowNxt; flowNxt = t; }
    }

    // context head (input = current flow, 2 -> 32 padded channels)
    {
      borderZero(flow16, h, w, 32);
      const int n = Bn * HW * 32;
      k_nchw_to_nhwc16<<<(n + 255) / 256, 256, 0, stream>>>(flowCur, flow16, Bn, 2, 32, h, w);
      const int cx_cinP[7] = {32, 128, 128, 128, 96, 64, 32};
      const int cx_cout[7] = {128, 128, 128, 96, 64, 32, 2};
      const _Float16* cur = flow16;
      _Float16* bufs[2] = {ping, pong};
      for (int j = 0; j < 6; ++j) {
        conv(cur, cxW[lvl][j], B_cx(lvl, j), bufs[j & 1], nullptr, nullptr,
             cx_cinP[j], h, w, cx_cout[j], padc(cx_cout[j]), 1, 1);
        cur = bufs[j & 1];
      }
      conv(cur, cxW[lvl][6], B_cx(lvl, 6), nullptr, flowCur, flowNxt,
           32, h, w, 2, 32, 1, 0);
      { float* t = flowCur; flowCur = flowNxt; flowNxt = t; }
    }
  }

  // ---- final 4x upsample, scale by 4 ----
  {
    const int n = Bn * 2 * 768 * 1024;
    k_upsample<<<(n + 255) / 256, 256, 0, stream>>>(flowCur, (float*)d_out,
                                                    Bn, 2, 192, 256, 768, 1024, 4.0f);
  }
}
